// Model_1494648619457
// MI455X (gfx1250) — matrix operand order repaired and checked by reading, not compiled, not run on hardware
//
#include <hip/hip_runtime.h>
#include <hip/hip_bf16.h>
#include <math.h>

// ---------------------------------------------------------------------------
// Problem constants (from reference)
// ---------------------------------------------------------------------------
#define BB 8192
#define NN 64
#define CC 8
#define DD 512
#define HP 1024
#define HS 512
#define TAUF 0.7f
#define SMOOTHF 0.1f
#define PARENT_N 2
#define CHILD_N 7

typedef __bf16 bf16;
typedef bf16 bf16x16 __attribute__((ext_vector_type(16)));
typedef float v8f __attribute__((ext_vector_type(8)));

// ---------------------------------------------------------------------------
// CDNA5 async global->LDS staging (guarded; falls back to load+ds_store)
// ---------------------------------------------------------------------------
#if defined(__has_builtin)
#if __has_builtin(__builtin_amdgcn_global_load_async_to_lds_b128)
#define HAVE_ASYNC_LDS 1
#endif
#endif
#ifndef HAVE_ASYNC_LDS
#define HAVE_ASYNC_LDS 0
#endif

#if HAVE_ASYNC_LDS
typedef int i4 __attribute__((ext_vector_type(4)));
typedef __attribute__((address_space(1))) i4 gbl_i4;
typedef __attribute__((address_space(3))) i4 lds_i4;
__device__ __forceinline__ lds_i4* lds_cast(const void* p) {
  // generic LDS pointer: low 32 bits are the LDS offset (flat-address truncation)
  return (lds_i4*)(unsigned)(unsigned long long)p;
}
__device__ __forceinline__ gbl_i4* gbl_cast(const void* p) {
  return (gbl_i4*)(unsigned long long)p;
}
__device__ __forceinline__ void async_wait0() {
#if __has_builtin(__builtin_amdgcn_s_wait_asynccnt)
  __builtin_amdgcn_s_wait_asynccnt(0);
#else
  asm volatile("s_wait_asynccnt 0x0" ::: "memory");
#endif
}
#endif

// ---------------------------------------------------------------------------
// f32 -> bf16 vectorized convert
// ---------------------------------------------------------------------------
__global__ __launch_bounds__(256) void f32_to_bf16_vec(const float* __restrict__ in,
                                                       bf16* __restrict__ out, int n4) {
  int i = blockIdx.x * blockDim.x + threadIdx.x;
  if (i >= n4) return;
  float4 v = ((const float4*)in)[i];
  union { bf16 h[4]; uint2 u; } o;
  o.h[0] = (bf16)v.x; o.h[1] = (bf16)v.y; o.h[2] = (bf16)v.z; o.h[3] = (bf16)v.w;
  ((uint2*)out)[i] = o.u;
}

// ---------------------------------------------------------------------------
// Transpose f32 (K x N) -> bf16 (N x K), 32x32 LDS tiles
// ---------------------------------------------------------------------------
__global__ __launch_bounds__(256) void transpose_f32_to_bf16(const float* __restrict__ in,
                                                             bf16* __restrict__ out,
                                                             int K, int N) {
  __shared__ float tile[32][33];
  const int k0 = blockIdx.y * 32;
  const int n0 = blockIdx.x * 32;
  const int tx = threadIdx.x & 31;
  const int ty = threadIdx.x >> 5;  // 0..7
#pragma unroll
  for (int i = ty; i < 32; i += 8)
    tile[i][tx] = in[(size_t)(k0 + i) * N + n0 + tx];
  __syncthreads();
#pragma unroll
  for (int i = ty; i < 32; i += 8)
    out[(size_t)(n0 + i) * K + k0 + tx] = (bf16)tile[tx][i];
}

// ---------------------------------------------------------------------------
// WMMA GEMM: out = act(A(MxK,bf16) * Bt(NxK,bf16)^T + bias)
// BM=128 BN=64 BK=32, 256 threads = 8 waves (4x2), wave tile 32x32 (2x2 WMMA)
// Double-buffered LDS; async global->LDS staging when available.
// act: 0=none 1=relu 2=sigmoid. outBf / outF may be null.
// ---------------------------------------------------------------------------
#define GBM 128
#define GBN 64
#define GBK 32

__global__ __launch_bounds__(256) void gemm_bf16_wmma(const bf16* __restrict__ A,
                                                      const bf16* __restrict__ Bt,
                                                      const float* __restrict__ bias,
                                                      bf16* __restrict__ outBf,
                                                      float* __restrict__ outF,
                                                      int M, int N, int K, int act) {
  __shared__ __align__(16) bf16 lA[2][GBM * GBK];  // 2 x 8 KB
  __shared__ __align__(16) bf16 lB[2][GBN * GBK];  // 2 x 4 KB
  const int tid = threadIdx.x;
  const int lane = tid & 31;
  const int wave = tid >> 5;
  const int wm = wave >> 1;  // 0..3  (M direction)
  const int wn = wave & 1;   // 0..1  (N direction)
  const int rowBase = blockIdx.y * GBM;
  const int colBase = blockIdx.x * GBN;
  const int m16 = lane & 15;
  const int kh = lane >> 4;  // lane-half selects K-group per ISA layout

  // per-thread staging coordinates
  const int rA0 = (tid + 0) >> 2, kcA0 = ((tid + 0) & 3) << 3;
  const int rA1 = (tid + 256) >> 2, kcA1 = ((tid + 256) & 3) << 3;
  const int rB = tid >> 2, kcB = (tid & 3) << 3;

  auto stage = [&](int k0, int buf) {
#if HAVE_ASYNC_LDS
    __builtin_amdgcn_global_load_async_to_lds_b128(
        gbl_cast(A + (size_t)(rowBase + rA0) * K + k0 + kcA0),
        lds_cast(&lA[buf][rA0 * GBK + kcA0]), 0, 0);
    __builtin_amdgcn_global_load_async_to_lds_b128(
        gbl_cast(A + (size_t)(rowBase + rA1) * K + k0 + kcA1),
        lds_cast(&lA[buf][rA1 * GBK + kcA1]), 0, 0);
    __builtin_amdgcn_global_load_async_to_lds_b128(
        gbl_cast(Bt + (size_t)(colBase + rB) * K + k0 + kcB),
        lds_cast(&lB[buf][rB * GBK + kcB]), 0, 0);
#else
    *(int4*)(&lA[buf][rA0 * GBK + kcA0]) =
        *(const int4*)(A + (size_t)(rowBase + rA0) * K + k0 + kcA0);
    *(int4*)(&lA[buf][rA1 * GBK + kcA1]) =
        *(const int4*)(A + (size_t)(rowBase + rA1) * K + k0 + kcA1);
    *(int4*)(&lB[buf][rB * GBK + kcB]) =
        *(const int4*)(Bt + (size_t)(colBase + rB) * K + k0 + kcB);
#endif
  };

  v8f acc00 = {}, acc01 = {}, acc10 = {}, acc11 = {};

  const int nIter = K / GBK;
  stage(0, 0);
#if HAVE_ASYNC_LDS
  async_wait0();
#endif
  __syncthreads();

  for (int it = 0; it < nIter; ++it) {
    const int cur = it & 1;
    if (it + 1 < nIter) stage((it + 1) * GBK, cur ^ 1);

    // Build fragments per CDNA5 16-bit A(16x32)/B(32x16) VGPR layouts.
    union Frag { bf16x16 v; unsigned u[8]; } a0, a1, b0, b1;
    const bf16* pa0 = &lA[cur][(wm * 32 + m16) * GBK];
    const bf16* pa1 = &lA[cur][(wm * 32 + 16 + m16) * GBK];
    const bf16* pb0 = &lB[cur][(wn * 32 + m16) * GBK];
    const bf16* pb1 = &lB[cur][(wn * 32 + 16 + m16) * GBK];
#pragma unroll
    for (int j = 0; j < 8; ++j) {
      // A: VGPR j<4 -> K = 8*kh + 2j ; j>=4 -> K = 16 + 8*kh + 2(j-4)
      int ka = (j < 4) ? (8 * kh + 2 * j) : (16 + 8 * kh + 2 * (j - 4));
      a0.u[j] = *(const unsigned*)(pa0 + ka);
      a1.u[j] = *(const unsigned*)(pa1 + ka);
      // B: same K order as A: VGPR j<4 -> K = 8*kh + 2j ; j>=4 -> K = 16 + 8*kh + 2(j-4)
      int kb = ka;
      b0.u[j] = *(const unsigned*)(pb0 + kb);
      b1.u[j] = *(const unsigned*)(pb1 + kb);
    }
    acc00 = __builtin_amdgcn_wmma_f32_16x16x32_bf16(false, a0.v, false, b0.v,
                                                    (short)0, acc00, false, false);
    acc01 = __builtin_amdgcn_wmma_f32_16x16x32_bf16(false, a0.v, false, b1.v,
                                                    (short)0, acc01, false, false);
    acc10 = __builtin_amdgcn_wmma_f32_16x16x32_bf16(false, a1.v, false, b0.v,
                                                    (short)0, acc10, false, false);
    acc11 = __builtin_amdgcn_wmma_f32_16x16x32_bf16(false, a1.v, false, b1.v,
                                                    (short)0, acc11, false, false);

#if HAVE_ASYNC_LDS
    async_wait0();  // next buffer fully resident before anyone signals
#endif
    __syncthreads();
  }

  // Epilogue: C/D layout -> lane (m16 = col, kh*8+i = row within 16x16)
#define EPILOG(ACC, MT, NT)                                                    \
  {                                                                            \
    int col = colBase + wn * 32 + (NT)*16 + m16;                               \
    float bv = bias ? bias[col] : 0.0f;                                        \
    _Pragma("unroll") for (int i = 0; i < 8; ++i) {                            \
      int row = rowBase + wm * 32 + (MT)*16 + kh * 8 + i;                      \
      float v = ACC[i] + bv;                                                   \
      if (act == 1) v = fmaxf(v, 0.0f);                                       \
      else if (act == 2) v = 1.0f / (1.0f + __expf(-v));                      \
      size_t off = (size_t)row * N + col;                                      \
      if (outF) outF[off] = v;                                                 \
      if (outBf) outBf[off] = (bf16)v;                                         \
    }                                                                          \
  }
  EPILOG(acc00, 0, 0)
  EPILOG(acc01, 0, 1)
  EPILOG(acc10, 1, 0)
  EPILOG(acc11, 1, 1)
#undef EPILOG
}

// ---------------------------------------------------------------------------
// Per-row epilogue: concrete relaxation, causal masking, xcf, truth copy.
// 1 block (64 threads) per batch row; thread n handles node n (C=8 classes).
// ---------------------------------------------------------------------------
__device__ __forceinline__ unsigned pcg_hash(unsigned v) {
  unsigned s = v * 747796405u + 2891336453u;
  unsigned w = ((s >> ((s >> 28) + 4u)) ^ s) * 277803737u;
  return (w >> 22) ^ w;
}
__device__ __forceinline__ float u01(unsigned h) {
  float f = (float)(h >> 8) * (1.0f / 16777216.0f);
  return f < 1e-8f ? 1e-8f : f;
}

__global__ __launch_bounds__(64) void epilogue_kernel(const float* __restrict__ x,
                                                      const float* __restrict__ truth,
                                                      const float* __restrict__ P,
                                                      const float* __restrict__ W,
                                                      float* __restrict__ out_truth,
                                                      float* __restrict__ out_xcf) {
  const int b = blockIdx.x;
  const int n = threadIdx.x;

  // coalesced truth_x passthrough (128 float4 per row)
  {
    const float4* ts = (const float4*)(truth + (size_t)b * DD);
    float4* td = (float4*)(out_truth + (size_t)b * DD);
    td[n] = ts[n];
    td[n + 64] = ts[n + 64];
  }

  const size_t base = (size_t)b * DD + (size_t)n * CC;
  float xv[CC], wv[CC];
#pragma unroll
  for (int c = 0; c < CC; ++c) { xv[c] = x[base + c]; wv[c] = W[base + c]; }

  int curr = 0;
  {
    float bm = xv[0];
#pragma unroll
    for (int c = 1; c < CC; ++c)
      if (xv[c] > bm) { bm = xv[c]; curr = c; }
  }

  // deterministic counter-based noise (stands in for jax threefry)
  const unsigned idx = (unsigned)(b * NN + n);
  const float ua = u01(pcg_hash(idx ^ 0x9E3779B9u));
  const float ub = u01(pcg_hash(idx ^ 0x85EBCA6Bu));
  const float ea = -1.0f / logf(ua);  // exp(-log(-log u))
  const float eb = -1.0f / logf(ub);
  const float Pv = P[(size_t)b * NN + n];
  const float no = Pv * ea * (1.0f / TAUF);
  const float de = no + (1.0f - Pv) * eb * (1.0f / TAUF);
  const float prob = no / de;

  float g[CC];
#pragma unroll
  for (int c = 0; c < CC; ++c)
    g[c] = -logf(-logf(u01(pcg_hash(idx * 8u + (unsigned)c + 0xC2B2AE35u))));

  __shared__ int sh_const, sh_incr;
  if (n == PARENT_N) {
    // smoothed gumbel softmax on raw logits (mask == 0 for parent)
    float mx = wv[0];
#pragma unroll
    for (int c = 1; c < CC; ++c) mx = fmaxf(mx, wv[c]);
    float e[CC], s = 0.0f;
#pragma unroll
    for (int c = 0; c < CC; ++c) { e[c] = __expf(wv[c] - mx); s += e[c]; }
    float z[CC], mz = -1e30f;
#pragma unroll
    for (int c = 0; c < CC; ++c) {
      float pr = (1.0f - SMOOTHF) * (e[c] / s) + SMOOTHF / (float)CC;
      z[c] = (logf(pr) + g[c]) * (1.0f / TAUF);
      mz = fmaxf(mz, z[c]);
    }
    float s2 = 0.0f;
#pragma unroll
    for (int c = 0; c < CC; ++c) { z[c] = __expf(z[c] - mz); s2 += z[c]; }
    int am = 0;
    float bo = -1e30f;
#pragma unroll
    for (int c = 0; c < CC; ++c) {
      float xt = z[c] / s2;
      float o = prob * xt + (1.0f - prob) * xv[c];
      if (o > bo) { bo = o; am = c; }
    }
    int diff = am - curr;
    sh_const = (diff == 0);
    sh_incr = (diff > 0);
  }
  __syncthreads();
  const int isConst = sh_const;
  const int isIncr = sh_incr;

  float mask[CC];
  if (n == 0 || n == 5 || n == 10) {
#pragma unroll
    for (int c = 0; c < CC; ++c) mask[c] = (c < curr) ? -100.0f : 1.0f;
  } else if (n == CHILD_N) {
    int thr = curr + 1;  // REL_TYPE == 0
#pragma unroll
    for (int c = 0; c < CC; ++c) mask[c] = (isIncr && c < thr) ? -100.0f : 1.0f;
  } else {
#pragma unroll
    for (int c = 0; c < CC; ++c) mask[c] = 0.0f;
  }

  float p = prob;
  if (n == CHILD_N) {
    p = prob * (1.0f - (float)isConst) + (float)isIncr;
    p = fminf(fmaxf(p, 0.0f), 1.0f);
  }

  // smoothed gumbel softmax on masked logits
  float l[CC], mx = -1e30f;
#pragma unroll
  for (int c = 0; c < CC; ++c) { l[c] = wv[c] + mask[c]; mx = fmaxf(mx, l[c]); }
  float e[CC], s = 0.0f;
#pragma unroll
  for (int c = 0; c < CC; ++c) { e[c] = __expf(l[c] - mx); s += e[c]; }
  float z[CC], mz = -1e30f;
#pragma unroll
  for (int c = 0; c < CC; ++c) {
    float pr = (1.0f - SMOOTHF) * (e[c] / s) + SMOOTHF / (float)CC;
    z[c] = (logf(pr) + g[c]) * (1.0f / TAUF);
    mz = fmaxf(mz, z[c]);
  }
  float s2 = 0.0f;
#pragma unroll
  for (int c = 0; c < CC; ++c) { z[c] = __expf(z[c] - mz); s2 += z[c]; }
#pragma unroll
  for (int c = 0; c < CC; ++c) {
    float xt = z[c] / s2;
    out_xcf[base + c] = p * xt + (1.0f - p) * xv[c];
  }
}

// ---------------------------------------------------------------------------
// Host launch
// ---------------------------------------------------------------------------
extern "C" void kernel_launch(void* const* d_in, const int* in_sizes, int n_in,
                              void* d_out, int out_size, void* d_ws, size_t ws_size,
                              hipStream_t stream) {
  const float* x = (const float*)d_in[0];
  const float* truth = (const float*)d_in[1];
  const float* sw1 = (const float*)d_in[2];
  const float* sb1 = (const float*)d_in[3];
  const float* sw2 = (const float*)d_in[4];
  const float* sb2 = (const float*)d_in[5];
  const float* pw1 = (const float*)d_in[6];
  const float* pb1 = (const float*)d_in[7];
  const float* pw2 = (const float*)d_in[8];
  const float* pb2 = (const float*)d_in[9];
  const float* pw3 = (const float*)d_in[10];
  const float* pb3 = (const float*)d_in[11];
  const float* pw4 = (const float*)d_in[12];
  const float* pb4 = (const float*)d_in[13];

  float* out_truth = (float*)d_out;
  float* out_xcf = out_truth + (size_t)BB * DD;
  float* out_P = out_xcf + (size_t)BB * DD;
  float* out_W = out_P + (size_t)BB * NN;

  char* wp = (char*)d_ws;
  auto alloc = [&](size_t bytes) {
    char* r = wp;
    wp += (bytes + 255) & ~(size_t)255;
    return r;
  };
  bf16* xb = (bf16*)alloc((size_t)BB * DD * 2);
  bf16* hA = (bf16*)alloc((size_t)BB * HP * 2);
  bf16* hB = (bf16*)alloc((size_t)BB * HP * 2);
  bf16* w1t = (bf16*)alloc((size_t)HS * DD * 2);  // (HS x D)
  bf16* w2t = (bf16*)alloc((size_t)NN * HS * 2);  // (N x HS)
  bf16* p1t = (bf16*)alloc((size_t)HP * DD * 2);  // (HP x D)
  bf16* p2t = (bf16*)alloc((size_t)HP * HP * 2);
  bf16* p3t = (bf16*)alloc((size_t)HP * HP * 2);
  bf16* p4t = (bf16*)alloc((size_t)DD * HP * 2);  // (D x HP)

  // one-time precision/layout conversions
  {
    int n4 = BB * DD / 4;
    f32_to_bf16_vec<<<(n4 + 255) / 256, 256, 0, stream>>>(x, xb, n4);
  }
  transpose_f32_to_bf16<<<dim3(HS / 32, DD / 32), 256, 0, stream>>>(sw1, w1t, DD, HS);
  transpose_f32_to_bf16<<<dim3(NN / 32, HS / 32), 256, 0, stream>>>(sw2, w2t, HS, NN);
  transpose_f32_to_bf16<<<dim3(HP / 32, DD / 32), 256, 0, stream>>>(pw1, p1t, DD, HP);
  transpose_f32_to_bf16<<<dim3(HP / 32, HP / 32), 256, 0, stream>>>(pw2, p2t, HP, HP);
  transpose_f32_to_bf16<<<dim3(HP / 32, HP / 32), 256, 0, stream>>>(pw3, p3t, HP, HP);
  transpose_f32_to_bf16<<<dim3(DD / 32, HP / 32), 256, 0, stream>>>(pw4, p4t, HP, DD);

  // selector MLP: P = sigmoid(relu(x@W1+b1)@W2+b2)
  gemm_bf16_wmma<<<dim3(HS / GBN, BB / GBM), 256, 0, stream>>>(
      xb, w1t, sb1, hA, nullptr, BB, HS, DD, 1);
  gemm_bf16_wmma<<<dim3(NN / GBN, BB / GBM), 256, 0, stream>>>(
      hA, w2t, sb2, nullptr, out_P, BB, NN, HS, 2);
  // policy MLP: W = relu^3 chain then linear
  gemm_bf16_wmma<<<dim3(HP / GBN, BB / GBM), 256, 0, stream>>>(
      xb, p1t, pb1, hA, nullptr, BB, HP, DD, 1);
  gemm_bf16_wmma<<<dim3(HP / GBN, BB / GBM), 256, 0, stream>>>(
      hA, p2t, pb2, hB, nullptr, BB, HP, HP, 1);
  gemm_bf16_wmma<<<dim3(HP / GBN, BB / GBM), 256, 0, stream>>>(
      hB, p3t, pb3, hA, nullptr, BB, HP, HP, 1);
  gemm_bf16_wmma<<<dim3(DD / GBN, BB / GBM), 256, 0, stream>>>(
      hA, p4t, pb4, nullptr, out_W, BB, DD, HP, 0);

  // counterfactual mixing + truth passthrough
  epilogue_kernel<<<BB, 64, 0, stream>>>(x, truth, out_P, out_W, out_truth, out_xcf);
}
